// EquivariantNodeFFN_78993038508799
// MI455X (gfx1250) — matrix operand order repaired and checked by reading, not compiled, not run on hardware
//
#include <hip/hip_runtime.h>
#include <cmath>

#define NCOL 480
#define EPSc 1e-8f
#define Z8 {0.f, 0.f, 0.f, 0.f, 0.f, 0.f, 0.f, 0.f}

typedef __bf16 bf16x16 __attribute__((ext_vector_type(16)));
typedef float  f32x8   __attribute__((ext_vector_type(8)));
typedef unsigned short u16x8 __attribute__((ext_vector_type(8)));

union FragU { bf16x16 v; u16x8 h[2]; };

__device__ __forceinline__ unsigned short f2bf(float f) {
  unsigned u = __builtin_bit_cast(unsigned, f);
  u += 0x7FFFu + ((u >> 16) & 1u);           // round-to-nearest-even
  return (unsigned short)(u >> 16);
}
__device__ __forceinline__ float bf2f(unsigned short h) {
  unsigned u = ((unsigned)h) << 16;
  return __builtin_bit_cast(float, u);
}
// fast sigmoid: v_exp_f32 + v_rcp_f32 (both TRANS, co-execute with XDL WMMA)
__device__ __forceinline__ float sigmoidf_(float x) {
  return __builtin_amdgcn_rcpf(1.0f + __expf(-x));
}
__device__ __forceinline__ float wred(float v) {
#pragma unroll
  for (int o = 16; o > 0; o >>= 1) v += __shfl_xor(v, o, 32);
  return v;
}

// A fragment (16x32 bf16, M x K): row-major ushort matrix, row stride ldk.
// lane<16: K = kbase+0..7 (v0-3), kbase+16..23 (v4-7); lane>=16: +8.
__device__ __forceinline__ bf16x16 loadA(const unsigned short* base, int ldk, int kbase, int lane) {
  const int row = lane & 15, kh = lane >> 4;
  const unsigned short* p = base + row * ldk + kbase + kh * 8;
  FragU f;
  f.h[0] = *(const u16x8*)(p);
  f.h[1] = *(const u16x8*)(p + 16);
  return f.v;
}
// B fragment (32x16 bf16, K x N) from pre-transposed weight Wt (N x K row-major).
// lane<16: col=lane, K = kbase+0..7 (v0-3), kbase+16..23 (v4-7); lane>=16: +8.
__device__ __forceinline__ bf16x16 loadB(const unsigned short* wt, int ldk, int nbase, int kbase, int lane) {
  const int col = lane & 15, kh = lane >> 4;
  const unsigned short* p = wt + (size_t)(nbase + col) * ldk + kbase + kh * 8;
  FragU f;
  f.h[0] = *(const u16x8*)(p);
  f.h[1] = *(const u16x8*)(p + 16);
  return f.v;
}
__device__ __forceinline__ f32x8 wmma_bf16(bf16x16 a, bf16x16 b, f32x8 c) {
  return __builtin_amdgcn_wmma_f32_16x16x32_bf16(false, a, false, b, (short)0, c, false, false);
}

#define SLICE_BYTES 39936
// per-wave slice layout (fits 2 workgroups per WGP: 2*4*39936 = 319488 <= 320KB):
//   [0,16384)      : zs  ushort 16x512  (silu output s; later reused as z-buffer)
//   [16384,28672)  : gs  ushort 16x384  (sigmoid gates)
//                    y0 staging (16x128) OVERLAID at 16384: consumed into regs
//                    at top of phase 2, before any gs store (per-wave LDS in-order)
//   [28672,39936)  : y12 ushort 16x352  (y1_m: cols 0..191, y2_m: cols 192..351)

__global__ __launch_bounds__(128) void ffn_kernel(
    const float* __restrict__ x,
    const float* __restrict__ nw0, const float* __restrict__ nb0,
    const float* __restrict__ nw1, const float* __restrict__ nw2,
    const float* __restrict__ b0,  const float* __restrict__ c0,
    const unsigned short* __restrict__ W0t, const unsigned short* __restrict__ W1t,
    const unsigned short* __restrict__ W2t, const unsigned short* __restrict__ V0t,
    const unsigned short* __restrict__ V1t, const unsigned short* __restrict__ V2t,
    const float* __restrict__ alpha_p,
    float* __restrict__ out)
{
  extern __shared__ char smem[];
  const int lane = threadIdx.x & 31;
  const int wave = threadIdx.x >> 5;
  const int row0 = (blockIdx.x * 4 + wave) * 16;
  char* my = smem + wave * SLICE_BYTES;
  unsigned short* zs  = (unsigned short*)my;
  unsigned short* gs  = (unsigned short*)(my + 16384);
  unsigned short* y0s = (unsigned short*)(my + 16384);   // overlaid with gs
  unsigned short* y12 = (unsigned short*)(my + 28672);

  const float tA = tanhf(alpha_p[0]);
  const int col = lane & 15, kh = lane >> 4;

  // ---- Phase 1: per-row stats (global -> regs) + normalized bf16 A-staging ----
  // Hoisted per-lane constants (independent of row):
  float w0v[4], b0v[4], w1v[6], w2v[5];
  int d1[6], d2[5];
#pragma unroll
  for (int i = 0; i < 4; ++i) { int k = lane + 32 * i; w0v[i] = nw0[k]; b0v[i] = nb0[k]; }
#pragma unroll
  for (int i = 0; i < 6; ++i) {
    int k = lane + 32 * i;                   // 0..191 : x1[u,m], k = u*3+m
    int u = k / 3, m = k - 3 * u;
    w1v[i] = nw1[u]; d1[i] = 64 * m + u;     // y12 col
  }
#pragma unroll
  for (int i = 0; i < 5; ++i) {
    int k = lane + 32 * i;                   // 0..159 : x2[u,m], k = u*5+m
    int u = k / 5, m = k - 5 * u;
    w2v[i] = nw2[u]; d2[i] = 192 + 32 * m + u;
  }

  for (int r = 0; r < 16; ++r) {
    const float* xr = x + (size_t)(row0 + r) * NCOL;
    float v0[4], v1[6], v2[5];
    float s0 = 0.f, q0 = 0.f, q1 = 0.f, q2 = 0.f;
#pragma unroll
    for (int i = 0; i < 4; ++i) { v0[i] = xr[lane + 32 * i];       s0 += v0[i]; q0 += v0[i] * v0[i]; }
#pragma unroll
    for (int i = 0; i < 6; ++i) { v1[i] = xr[128 + lane + 32 * i]; q1 += v1[i] * v1[i]; }
#pragma unroll
    for (int i = 0; i < 5; ++i) { v2[i] = xr[320 + lane + 32 * i]; q2 += v2[i] * v2[i]; }
    s0 = wred(s0); q0 = wred(q0); q1 = wred(q1); q2 = wred(q2);
    const float mu  = s0 * (1.f / 128.f);
    const float rs0 = rsqrtf(fmaxf(q0 * (1.f / 128.f) - mu * mu, 0.f) + EPSc);
    const float r12 = rsqrtf(0.5f * (q1 * (1.f / 192.f) + q2 * (1.f / 160.f)) + EPSc);
#pragma unroll
    for (int i = 0; i < 4; ++i)
      y0s[r * 128 + lane + 32 * i] = f2bf((v0[i] - mu) * rs0 * w0v[i] + b0v[i]);
#pragma unroll
    for (int i = 0; i < 6; ++i)
      y12[r * 352 + d1[i]] = f2bf(v1[i] * r12 * w1v[i]);
#pragma unroll
    for (int i = 0; i < 5; ++i)
      y12[r * 352 + d2[i]] = f2bf(v2[i] * r12 * w2v[i]);
  }
  __syncthreads();

  // ---- Phase 2: h0 = y0 @ W0t + b0 ; s = silu -> zs, g = sigmoid -> gs ----
  // Dual N-tiles per iteration -> 2 independent WMMA chains sharing A.
  {
    bf16x16 a0[4];
#pragma unroll
    for (int kt = 0; kt < 4; ++kt) a0[kt] = loadA(y0s, 128, kt * 32, lane);
    for (int nt = 0; nt < 56; nt += 2) {
      if (nt < 54)
        __builtin_prefetch(W0t + (size_t)((nt + 2) * 16 + col) * 128, 0, 1);
      f32x8 acc[2] = {Z8, Z8};
#pragma unroll
      for (int kt = 0; kt < 4; ++kt) {
        bf16x16 bA = loadB(W0t, 128, nt * 16, kt * 32, lane);
        bf16x16 bB = loadB(W0t, 128, nt * 16 + 16, kt * 32, lane);
        acc[0] = wmma_bf16(a0[kt], bA, acc[0]);
        acc[1] = wmma_bf16(a0[kt], bB, acc[1]);
      }
      if (nt < 32) {
#pragma unroll
        for (int t = 0; t < 2; ++t) {
          const int c = (nt + t) * 16 + col;
          const float bias = b0[c];
#pragma unroll
          for (int j = 0; j < 8; ++j) {
            float h = acc[t][j] + bias;
            zs[(j + 8 * kh) * 512 + c] = f2bf(h * sigmoidf_(h));
          }
        }
      } else {
#pragma unroll
        for (int t = 0; t < 2; ++t) {
          const int c = (nt + t) * 16 + col;
          const float bias = b0[c];
#pragma unroll
          for (int j = 0; j < 8; ++j) {
            float h = acc[t][j] + bias;
            gs[(j + 8 * kh) * 384 + (c - 512)] = f2bf(sigmoidf_(h));
          }
        }
      }
    }
  }
  __syncthreads();

  // ---- Phase 3: o0 = s @ V0t + c0 ; residual ----
  // Dual-N x even/odd-K split -> 4 independent WMMA chains of depth 8.
  for (int np = 0; np < 4; ++np) {
    f32x8 acc[2][2] = {{Z8, Z8}, {Z8, Z8}};
#pragma unroll
    for (int kt = 0; kt < 16; kt += 2) {
      bf16x16 aE = loadA(zs, 512, kt * 32, lane);
      bf16x16 aO = loadA(zs, 512, kt * 32 + 32, lane);
      acc[0][0] = wmma_bf16(aE, loadB(V0t, 512, np * 32,      kt * 32,      lane), acc[0][0]);
      acc[1][0] = wmma_bf16(aE, loadB(V0t, 512, np * 32 + 16, kt * 32,      lane), acc[1][0]);
      acc[0][1] = wmma_bf16(aO, loadB(V0t, 512, np * 32,      kt * 32 + 32, lane), acc[0][1]);
      acc[1][1] = wmma_bf16(aO, loadB(V0t, 512, np * 32 + 16, kt * 32 + 32, lane), acc[1][1]);
    }
#pragma unroll
    for (int t = 0; t < 2; ++t) {
      const f32x8 a = acc[t][0] + acc[t][1];
      const int c = np * 32 + t * 16 + col;
      const float bias = c0[c];
#pragma unroll
      for (int j = 0; j < 8; ++j) {
        const size_t off = (size_t)(row0 + j + 8 * kh) * NCOL + c;
        out[off] = x[off] + tA * (a[j] + bias);
      }
    }
  }
  __syncthreads();

  // ---- Phase 4: per m: h1_m = y1_m @ W1t, gate by g[:, :256], o1_m = z @ V1t ----
  for (int m = 0; m < 3; ++m) {
    bf16x16 a1[2];
#pragma unroll
    for (int kt = 0; kt < 2; ++kt) a1[kt] = loadA(y12, 352, 64 * m + kt * 32, lane);
    for (int np = 0; np < 8; ++np) {           // 16 N-tiles as 8 pairs
      f32x8 acc[2] = {Z8, Z8};
#pragma unroll
      for (int kt = 0; kt < 2; ++kt) {
        acc[0] = wmma_bf16(a1[kt], loadB(W1t, 64, np * 32, kt * 32, lane), acc[0]);
        acc[1] = wmma_bf16(a1[kt], loadB(W1t, 64, np * 32 + 16, kt * 32, lane), acc[1]);
      }
#pragma unroll
      for (int t = 0; t < 2; ++t) {
        const int c = np * 32 + t * 16 + col;
#pragma unroll
        for (int j = 0; j < 8; ++j) {
          const int rr = j + 8 * kh;
          zs[rr * 256 + c] = f2bf(acc[t][j] * bf2f(gs[rr * 384 + c]));
        }
      }
    }
    __syncthreads();
    for (int np = 0; np < 2; ++np) {           // 4 N-tiles as 2 pairs, even/odd K split
      f32x8 acc[2][2] = {{Z8, Z8}, {Z8, Z8}};
#pragma unroll
      for (int kt = 0; kt < 8; kt += 2) {
        bf16x16 aE = loadA(zs, 256, kt * 32, lane);
        bf16x16 aO = loadA(zs, 256, kt * 32 + 32, lane);
        acc[0][0] = wmma_bf16(aE, loadB(V1t, 256, np * 32,      kt * 32,      lane), acc[0][0]);
        acc[1][0] = wmma_bf16(aE, loadB(V1t, 256, np * 32 + 16, kt * 32,      lane), acc[1][0]);
        acc[0][1] = wmma_bf16(aO, loadB(V1t, 256, np * 32,      kt * 32 + 32, lane), acc[0][1]);
        acc[1][1] = wmma_bf16(aO, loadB(V1t, 256, np * 32 + 16, kt * 32 + 32, lane), acc[1][1]);
      }
#pragma unroll
      for (int t = 0; t < 2; ++t) {
        const f32x8 a = acc[t][0] + acc[t][1];
        const int c = np * 32 + t * 16 + col;
#pragma unroll
        for (int j = 0; j < 8; ++j) {
          const size_t off = (size_t)(row0 + j + 8 * kh) * NCOL + 128 + c * 3 + m;
          out[off] = x[off] + tA * a[j];
        }
      }
    }
    __syncthreads();
  }

  // ---- Phase 5: per m: h2_m = y2_m @ W2t, gate by g[:, 256:], o2_m = z @ V2t ----
  for (int m = 0; m < 5; ++m) {
    const bf16x16 a2 = loadA(y12, 352, 192 + 32 * m, lane);
    for (int np = 0; np < 4; ++np) {           // 8 N-tiles as 4 pairs
      f32x8 acc[2] = {Z8, Z8};
      acc[0] = wmma_bf16(a2, loadB(W2t, 32, np * 32, 0, lane), acc[0]);
      acc[1] = wmma_bf16(a2, loadB(W2t, 32, np * 32 + 16, 0, lane), acc[1]);
#pragma unroll
      for (int t = 0; t < 2; ++t) {
        const int c = np * 32 + t * 16 + col;
#pragma unroll
        for (int j = 0; j < 8; ++j) {
          const int rr = j + 8 * kh;
          zs[rr * 128 + c] = f2bf(acc[t][j] * bf2f(gs[rr * 384 + 256 + c]));
        }
      }
    }
    __syncthreads();
    {                                          // 2 N-tiles as 1 pair, even/odd K split
      f32x8 acc[2][2] = {{Z8, Z8}, {Z8, Z8}};
#pragma unroll
      for (int kt = 0; kt < 4; kt += 2) {
        bf16x16 aE = loadA(zs, 128, kt * 32, lane);
        bf16x16 aO = loadA(zs, 128, kt * 32 + 32, lane);
        acc[0][0] = wmma_bf16(aE, loadB(V2t, 128, 0,  kt * 32,      lane), acc[0][0]);
        acc[1][0] = wmma_bf16(aE, loadB(V2t, 128, 16, kt * 32,      lane), acc[1][0]);
        acc[0][1] = wmma_bf16(aO, loadB(V2t, 128, 0,  kt * 32 + 32, lane), acc[0][1]);
        acc[1][1] = wmma_bf16(aO, loadB(V2t, 128, 16, kt * 32 + 32, lane), acc[1][1]);
      }
#pragma unroll
      for (int t = 0; t < 2; ++t) {
        const f32x8 a = acc[t][0] + acc[t][1];
        const int c = t * 16 + col;
#pragma unroll
        for (int j = 0; j < 8; ++j) {
          const size_t off = (size_t)(row0 + j + 8 * kh) * NCOL + 320 + c * 5 + m;
          out[off] = x[off] + tA * a[j];
        }
      }
    }
    __syncthreads();
  }
}

// One-time weight prep: W (K x N f32, row-major) -> Wt (N x K bf16, row-major), scaled.
__global__ void prep_transpose_kernel(const float* __restrict__ src, unsigned short* __restrict__ dst,
                                      int K, int N, float scale) {
  int idx = blockIdx.x * blockDim.x + threadIdx.x;
  if (idx >= K * N) return;
  int n = idx / K, k = idx - n * K;
  dst[idx] = f2bf(src[(size_t)k * N + n] * scale);
}

extern "C" void kernel_launch(void* const* d_in, const int* in_sizes, int n_in,
                              void* d_out, int out_size, void* d_ws, size_t ws_size,
                              hipStream_t stream) {
  (void)n_in; (void)out_size; (void)ws_size;
  const float* x     = (const float*)d_in[0];
  const float* nw0   = (const float*)d_in[1];
  const float* nb0   = (const float*)d_in[2];
  const float* nw1   = (const float*)d_in[3];
  const float* nw2   = (const float*)d_in[4];
  const float* W0    = (const float*)d_in[5];
  const float* b0    = (const float*)d_in[6];
  const float* W1    = (const float*)d_in[7];
  const float* W2    = (const float*)d_in[8];
  const float* V0    = (const float*)d_in[9];
  const float* c0    = (const float*)d_in[10];
  const float* V1    = (const float*)d_in[11];
  const float* V2    = (const float*)d_in[12];
  const float* alpha = (const float*)d_in[13];
  float* out = (float*)d_out;

  unsigned short* ws  = (unsigned short*)d_ws;
  unsigned short* W0t = ws;                     // 896*128
  unsigned short* V0t = W0t + 896 * 128;        // 128*512
  unsigned short* W1t = V0t + 128 * 512;        // 256*64
  unsigned short* V1t = W1t + 256 * 64;         // 64*256
  unsigned short* W2t = V1t + 64 * 256;         // 128*32
  unsigned short* V2t = W2t + 128 * 32;         // 32*128

  auto prep = [&](const float* src, unsigned short* dst, int K, int N, float scale) {
    int total = K * N;
    prep_transpose_kernel<<<(total + 255) / 256, 256, 0, stream>>>(src, dst, K, N, scale);
  };
  prep(W0, W0t, 128, 896, 1.f / sqrtf(128.f));  // /S0
  prep(V0, V0t, 512, 128, 1.f / sqrtf(512.f));  // /T0
  prep(W1, W1t,  64, 256, 1.f / sqrtf(64.f));   // /S1
  prep(V1, V1t, 256,  64, 1.f / sqrtf(256.f));  // /T1
  prep(W2, W2t,  32, 128, 1.f / sqrtf(32.f));   // /S2
  prep(V2, V2t, 128,  32, 1.f / sqrtf(128.f));  // /T2

  const int nRows  = in_sizes[0] / NCOL;        // 65536
  const int blocks = nRows / 64;                // 4 waves x 16 rows per block
  ffn_kernel<<<blocks, 128, 4 * SLICE_BYTES, stream>>>(
      x, nw0, nb0, nw1, nw2, b0, c0,
      W0t, W1t, W2t, V0t, V1t, V2t, alpha, out);
}
